// Model_74921409511664
// MI455X (gfx1250) — compile-verified, matrix operand order checked by reading, not run on hardware
//
#include <hip/hip_runtime.h>
#include <hip/hip_bf16.h>

// ---------------------------------------------------------------------------
// Sizes / tiling
// ---------------------------------------------------------------------------
#define DC 300          // logical feature dim
#define DP 320          // padded feature dim (10 K-chunks of 32)
#define TM 128          // GEMM rows per workgroup
#define TN 64           // GEMM cols per workgroup
#define TK 32           // WMMA K per step
#define LDAS 40         // LDS stride (bf16) for A tile: 80B rows, 16B aligned, bank-conflict free
#define LDBS 328        // LDS stride (bf16) for B strip: 656B rows, 16B aligned
#define NGRAPH 1024

typedef __attribute__((ext_vector_type(16))) __bf16 v16bf;
typedef __attribute__((ext_vector_type(8)))  float  v8f;

union FragAB { v16bf v; uint4 q[2]; };

__device__ __forceinline__ __bf16 f2bf(float f) {
  union { float f; unsigned u; } a; a.f = f;
  unsigned u = a.u;
  u += 0x7fffu + ((u >> 16) & 1u);          // round-to-nearest-even
  union { unsigned short s; __bf16 b; } r;
  r.s = (unsigned short)(u >> 16);
  return r.b;
}

// ---------------------------------------------------------------------------
// WMMA GEMM:  C[M x Ncols] = alpha * (A[M x 320] @ B[Ncols x 320]^T) + bias
// A, B are bf16 row-major.  LDS staged with GLOBAL_LOAD_ASYNC_TO_LDS_B128
// (ASYNCcnt-tracked), consumed after s_wait_asynccnt 0 + workgroup barrier.
// 256 threads = 8 waves; wave w owns rows [w*16, w*16+16) x TN cols.
// Per K-chunk each wave issues 4 x v_wmma_f32_16x16x32_bf16 -> 40 WMMA / wave.
// ---------------------------------------------------------------------------
__global__ __launch_bounds__(256)
void gemm_wmma_kernel(const __bf16* __restrict__ A, int lda,
                      const __bf16* __restrict__ B, int ldb,
                      float* __restrict__ C, int ldc,
                      const float* __restrict__ bias, float alpha,
                      int M, int Ncols)
{
  __shared__ __attribute__((aligned(16))) __bf16 As[TM * LDAS];   // 10 KB
  __shared__ __attribute__((aligned(16))) __bf16 Bs[TN * LDBS];   // 41 KB

  const int tid = threadIdx.x;
  const int m0  = blockIdx.x * TM;
  const int n0  = blockIdx.y * TN;

  const unsigned asBase = (unsigned)(unsigned long long)(uintptr_t)As;
  const unsigned bsBase = (unsigned)(unsigned long long)(uintptr_t)Bs;
  const uint4 z4 = {0u, 0u, 0u, 0u};

  // Stage the full K-strip of B (TN rows x 320 bf16) once: 40 x b128 per row.
  for (int i = tid; i < TN * (DP / 8); i += 256) {
    int r = i / (DP / 8), c = i % (DP / 8);
    int gn = n0 + r;
    unsigned lo = bsBase + (unsigned)(r * LDBS * 2 + c * 16);
    if (gn < Ncols) {
      unsigned long long ga =
          (unsigned long long)(uintptr_t)(B + (size_t)gn * ldb + c * 8);
      asm volatile("global_load_async_to_lds_b128 %0, %1, off"
                   :: "v"(lo), "v"(ga) : "memory");
    } else {
      *(uint4*)&Bs[r * LDBS + c * 8] = z4;
    }
  }

  v8f zero8 = {0.f, 0.f, 0.f, 0.f, 0.f, 0.f, 0.f, 0.f};
  v8f acc[4];
  acc[0] = zero8; acc[1] = zero8; acc[2] = zero8; acc[3] = zero8;

  const int wave  = tid >> 5;
  const int lane  = tid & 31;
  const int arow  = wave * 16 + (lane & 15);
  const int kbase = (lane >> 4) << 3;       // 0 or 8 (16-bit A/B VGPR layout split)

  for (int kk = 0; kk < DP; kk += TK) {
    // Stage A chunk (TM x 32 bf16): 4 x b128 per row, 512 ops / 256 threads.
#pragma unroll
    for (int j = 0; j < 2; ++j) {
      int i = tid + j * 256;
      int r = i >> 2, c = i & 3;
      int gm = m0 + r;
      unsigned lo = asBase + (unsigned)(r * LDAS * 2 + c * 16);
      if (gm < M) {
        unsigned long long ga =
            (unsigned long long)(uintptr_t)(A + (size_t)gm * lda + kk + c * 8);
        asm volatile("global_load_async_to_lds_b128 %0, %1, off"
                     :: "v"(lo), "v"(ga) : "memory");
      } else {
        *(uint4*)&As[r * LDAS + c * 8] = z4;
      }
    }
    asm volatile("s_wait_asynccnt 0x0" ::: "memory");
    __syncthreads();

    FragAB a;
    a.q[0] = *(const uint4*)&As[arow * LDAS + kbase];
    a.q[1] = *(const uint4*)&As[arow * LDAS + 16 + kbase];
#pragma unroll
    for (int t = 0; t < 4; ++t) {
      FragAB b;
      const int brow = t * 16 + (lane & 15);
      b.q[0] = *(const uint4*)&Bs[brow * LDBS + kk + kbase];
      b.q[1] = *(const uint4*)&Bs[brow * LDBS + kk + 16 + kbase];
      acc[t] = __builtin_amdgcn_wmma_f32_16x16x32_bf16(
          false, a.v, false, b.v, (short)0, acc[t], false, false);
    }
    __syncthreads();   // protect As before next stage overwrites it
  }

  // C/D layout: VGPR vi, lanes<16 -> M = vi, lanes>=16 -> M = vi+8; N = lane&15.
  const int mb = m0 + wave * 16 + ((lane >> 4) << 3);
  const int nb = n0 + (lane & 15);
#pragma unroll
  for (int t = 0; t < 4; ++t) {
    int n = nb + t * 16;
    if (n >= Ncols) continue;
    float bv = bias ? bias[n] : 0.0f;
#pragma unroll
    for (int vi = 0; vi < 8; ++vi) {
      int m = mb + vi;
      if (m < M) C[(size_t)m * ldc + n] = acc[t][vi] * alpha + bv;
    }
  }
}

// ---------------------------------------------------------------------------
// Conversion / elementwise / graph kernels
// ---------------------------------------------------------------------------
__global__ void f32tobf16_kernel(const float* __restrict__ src,
                                 __bf16* __restrict__ dst, long n8) {
  long i = (long)blockIdx.x * blockDim.x + threadIdx.x;
  if (i >= n8) return;
  float4 a = *(const float4*)(src + i * 8);
  float4 b = *(const float4*)(src + i * 8 + 4);
  union { __bf16 h[8]; uint4 q; } u;
  u.h[0] = f2bf(a.x); u.h[1] = f2bf(a.y); u.h[2] = f2bf(a.z); u.h[3] = f2bf(a.w);
  u.h[4] = f2bf(b.x); u.h[5] = f2bf(b.y); u.h[6] = f2bf(b.z); u.h[7] = f2bf(b.w);
  *(uint4*)(dst + i * 8) = u.q;
}

__global__ void fill_kernel(float* __restrict__ p, float v, long n) {
  long i = (long)blockIdx.x * blockDim.x + threadIdx.x;
  if (i < n) p[i] = v;
}

__global__ void pad2d_kernel(const float* __restrict__ src, float* __restrict__ dst,
                             int sr, int sc, int dr, int dc, long total) {
  long i = (long)blockIdx.x * blockDim.x + threadIdx.x;
  if (i >= total) return;
  int per = dr * dc;
  int w = (int)(i / per);
  int rem = (int)(i % per);
  int r = rem / dc, c = rem % dc;
  dst[i] = (r < sr && c < sc) ? src[(size_t)w * sr * sc + (size_t)r * sc + c] : 0.0f;
}

__global__ void pad2d_bf16_kernel(const float* __restrict__ src, __bf16* __restrict__ dst,
                                  int sr, int sc, int dr, int dc, long total) {
  long i = (long)blockIdx.x * blockDim.x + threadIdx.x;
  if (i >= total) return;
  int per = dr * dc;
  int w = (int)(i / per);
  int rem = (int)(i % per);
  int r = rem / dc, c = rem % dc;
  dst[i] = (r < sr && c < sc) ? f2bf(src[(size_t)w * sr * sc + (size_t)r * sc + c])
                              : f2bf(0.0f);
}

__global__ void embed_kernel(const int* __restrict__ x, const float* __restrict__ ae1,
                             const float* __restrict__ ae2, float* __restrict__ h, int N) {
  long idx = (long)blockIdx.x * blockDim.x + threadIdx.x;
  if (idx >= (long)N * (DP / 4)) return;
  int r = (int)(idx / (DP / 4)), c4 = (int)(idx % (DP / 4));
  float4 o = make_float4(0.f, 0.f, 0.f, 0.f);
  if (c4 < DC / 4) {
    int a = x[2 * r], b = x[2 * r + 1];
    float4 v1 = *(const float4*)(ae1 + (size_t)a * DC + c4 * 4);
    float4 v2 = *(const float4*)(ae2 + (size_t)b * DC + c4 * 4);
    o.x = v1.x + v2.x; o.y = v1.y + v2.y; o.z = v1.z + v2.z; o.w = v1.w + v2.w;
  }
  *(float4*)(h + (size_t)r * DP + c4 * 4) = o;
}

__global__ void deg_kernel(const int* __restrict__ ei, float* __restrict__ deg, int E) {
  int e = blockIdx.x * blockDim.x + threadIdx.x;
  if (e < E) atomicAdd(&deg[ei[e]], 1.0f);   // ei row 0 = src
}

__global__ void rsqrt_kernel(float* __restrict__ d, int N) {
  int i = blockIdx.x * blockDim.x + threadIdx.x;
  if (i < N) d[i] = rsqrtf(d[i]);            // deg >= 1 always (self-loop)
}

// out[n] = dinv[n]^2 * (hlin[n] + ee1[4] + ee2[0])   (self-loop term; pads = 0)
__global__ void self_init_kernel(const float* __restrict__ hlin, const float* __restrict__ se1,
                                 const float* __restrict__ se2, const float* __restrict__ dinv,
                                 float* __restrict__ out, int N) {
  long idx = (long)blockIdx.x * blockDim.x + threadIdx.x;
  if (idx >= (long)N * (DP / 4)) return;
  int r = (int)(idx / (DP / 4)), c4 = (int)(idx % (DP / 4));
  float4 o = make_float4(0.f, 0.f, 0.f, 0.f);
  if (c4 < DC / 4) {
    float nm = dinv[r] * dinv[r];
    float4 v = *(const float4*)(hlin + (size_t)r * DP + c4 * 4);
    int c = c4 * 4;
    o.x = nm * (v.x + se1[c + 0] + se2[c + 0]);
    o.y = nm * (v.y + se1[c + 1] + se2[c + 1]);
    o.z = nm * (v.z + se1[c + 2] + se2[c + 2]);
    o.w = nm * (v.w + se1[c + 3] + se2[c + 3]);
  }
  *(float4*)(out + (size_t)r * DP + c4 * 4) = o;
}

// One wave (32 lanes) per edge: out[dst] += norm * (hlin[src] + ee1[bt] + ee2[bd])
__global__ void edge_msg_kernel(const int* __restrict__ ei, const int* __restrict__ ea,
                                const float* __restrict__ hlin, const float* __restrict__ ee1,
                                const float* __restrict__ ee2, const float* __restrict__ dinv,
                                float* __restrict__ out, int E) {
  long gid = (long)blockIdx.x * blockDim.x + threadIdx.x;
  int e = (int)(gid >> 5), lane = (int)(gid & 31);
  if (e >= E) return;
  int src = ei[e], dst = ei[E + e];
  int bt = ea[2 * e], bd = ea[2 * e + 1];
  float nm = dinv[src] * dinv[dst];
  const float* hs = hlin + (size_t)src * DP;
  const float* b1 = ee1 + (size_t)bt * DC;
  const float* b2 = ee2 + (size_t)bd * DC;
  float* od = out + (size_t)dst * DP;
  for (int c = lane; c < DC; c += 32)
    atomicAdd(&od[c], nm * (hs[c] + b1[c] + b2[c]));
}

__global__ void colstat_kernel(const float* __restrict__ h, float* __restrict__ sums,
                               float* __restrict__ sqs, int N) {
  int col = blockIdx.x;
  __shared__ float ls[256], lq[256];
  float s = 0.f, q = 0.f;
  for (int r = blockIdx.y * 256 + threadIdx.x; r < N; r += gridDim.y * 256) {
    float v = h[(size_t)r * DP + col];
    s += v; q += v * v;
  }
  ls[threadIdx.x] = s; lq[threadIdx.x] = q;
  __syncthreads();
  for (int o = 128; o > 0; o >>= 1) {
    if (threadIdx.x < o) { ls[threadIdx.x] += ls[threadIdx.x + o]; lq[threadIdx.x] += lq[threadIdx.x + o]; }
    __syncthreads();
  }
  if (threadIdx.x == 0) { atomicAdd(&sums[col], ls[0]); atomicAdd(&sqs[col], lq[0]); }
}

__global__ void bn_apply_kernel(float* __restrict__ h, const float* __restrict__ sums,
                                const float* __restrict__ sqs, const float* __restrict__ gamma,
                                const float* __restrict__ beta, int N, int relu) {
  long idx = (long)blockIdx.x * blockDim.x + threadIdx.x;
  if (idx >= (long)N * (DC / 4)) return;
  int r = (int)(idx / (DC / 4)), c4 = (int)(idx % (DC / 4));
  float4 v = *(float4*)(h + (size_t)r * DP + c4 * 4);
  float* vp = &v.x;
  float invN = 1.0f / (float)N;
#pragma unroll
  for (int i = 0; i < 4; ++i) {
    int c = c4 * 4 + i;
    float mean = sums[c] * invN;
    float var = sqs[c] * invN - mean * mean;
    float sc = gamma[c] * rsqrtf(var + 1e-5f);
    float y = sc * (vp[i] - mean) + beta[c];
    vp[i] = relu ? fmaxf(y, 0.f) : y;
  }
  *(float4*)(h + (size_t)r * DP + c4 * 4) = v;
}

__global__ void pool_scatter_kernel(const float* __restrict__ h, const int* __restrict__ ids,
                                    float* __restrict__ s, float* __restrict__ cnt, int Nrows) {
  long gid = (long)blockIdx.x * blockDim.x + threadIdx.x;
  int r = (int)(gid >> 5), lane = (int)(gid & 31);
  if (r >= Nrows) return;
  int g = ids[r];
  const float* hr = h + (size_t)r * DP;
  float* sg = s + (size_t)g * DP;
  for (int c = lane; c < DC; c += 32) atomicAdd(&sg[c], hr[c]);
  if (lane == 0) atomicAdd(&cnt[g], 1.0f);
}

__global__ void pool_div_kernel(float* __restrict__ s, const float* __restrict__ cnt, int M) {
  long idx = (long)blockIdx.x * blockDim.x + threadIdx.x;
  if (idx >= (long)M * (DP / 4)) return;
  int r = (int)(idx / (DP / 4)), c4 = (int)(idx % (DP / 4));
  float inv = 1.0f / fmaxf(cnt[r], 1.0f);
  float4 v = *(float4*)(s + (size_t)r * DP + c4 * 4);
  v.x *= inv; v.y *= inv; v.z *= inv; v.w *= inv;
  *(float4*)(s + (size_t)r * DP + c4 * 4) = v;
}

__global__ void mask_kernel(float* __restrict__ j, const float* __restrict__ mask,
                            const float* __restrict__ memb, int M) {
  long idx = (long)blockIdx.x * blockDim.x + threadIdx.x;
  if (idx >= (long)M * (DC / 4)) return;
  int r = (int)(idx / (DC / 4)), c4 = (int)(idx % (DC / 4));
  if (mask[r] > 0.5f)
    *(float4*)(j + (size_t)r * DP + c4 * 4) = *(const float4*)(memb + c4 * 4);
}

// Self-loop-only GCNConv tail: h += ee1[4] + ee2[0]; optional ReLU (norm==1).
__global__ void addse_kernel(float* __restrict__ h, const float* __restrict__ se1,
                             const float* __restrict__ se2, int M, int relu) {
  long idx = (long)blockIdx.x * blockDim.x + threadIdx.x;
  if (idx >= (long)M * (DC / 4)) return;
  int r = (int)(idx / (DC / 4)), c4 = (int)(idx % (DC / 4));
  float4 v = *(float4*)(h + (size_t)r * DP + c4 * 4);
  int c = c4 * 4;
  v.x += se1[c + 0] + se2[c + 0];
  v.y += se1[c + 1] + se2[c + 1];
  v.z += se1[c + 2] + se2[c + 2];
  v.w += se1[c + 3] + se2[c + 3];
  if (relu) { v.x = fmaxf(v.x, 0.f); v.y = fmaxf(v.y, 0.f); v.z = fmaxf(v.z, 0.f); v.w = fmaxf(v.w, 0.f); }
  *(float4*)(h + (size_t)r * DP + c4 * 4) = v;
}

__global__ void l2norm_kernel(const float* __restrict__ in, float* __restrict__ out, int M) {
  long gid = (long)blockIdx.x * blockDim.x + threadIdx.x;
  int r = (int)(gid >> 5), lane = (int)(gid & 31);
  if (r >= M) return;
  float s = 0.f;
  for (int c = lane; c < DC; c += 32) { float v = in[(size_t)r * DP + c]; s += v * v; }
  for (int m = 16; m > 0; m >>= 1) s += __shfl_xor(s, m, 32);
  float sc = 1.0f / fmaxf(sqrtf(s), 1e-12f);
  for (int c = lane; c < DP; c += 32)
    out[(size_t)r * DP + c] = (c < DC) ? in[(size_t)r * DP + c] * sc : 0.f;
}

// ---------------------------------------------------------------------------
// Orchestration
// ---------------------------------------------------------------------------
extern "C" void kernel_launch(void* const* d_in, const int* in_sizes, int n_in,
                              void* d_out, int out_size, void* d_ws, size_t ws_size,
                              hipStream_t stream) {
  (void)n_in; (void)out_size; (void)ws_size;
  const int*   batch_x   = (const int*)d_in[0];
  const int*   batch_ei  = (const int*)d_in[1];
  const int*   batch_ea  = (const int*)d_in[2];
  const int*   batch_gid = (const int*)d_in[3];
  const int*   frag_x    = (const int*)d_in[4];
  const int*   frag_ei   = (const int*)d_in[5];
  const int*   frag_ea   = (const int*)d_in[6];
  const int*   frag_jid  = (const int*)d_in[7];
  const int*   jct_gid   = (const int*)d_in[8];
  const float* jct_mask  = (const float*)d_in[9];
  const float* ae1       = (const float*)d_in[10];
  const float* ae2       = (const float*)d_in[11];
  const float* gnn_W     = (const float*)d_in[12];
  const float* gnn_b     = (const float*)d_in[13];
  const float* gnn_ee1   = (const float*)d_in[14];
  const float* gnn_ee2   = (const float*)d_in[15];
  const float* bn_gamma  = (const float*)d_in[16];
  const float* bn_beta   = (const float*)d_in[17];
  const float* jct_Wsrc  = (const float*)d_in[18];
  const float* jct_bsrc  = (const float*)d_in[19];
  const float* jct_ee1   = (const float*)d_in[20];
  const float* jct_ee2   = (const float*)d_in[21];
  const float* mask_emb  = (const float*)d_in[22];

  const int N = in_sizes[0] / 2;
  const int E = in_sizes[1] / 2;
  const int J = in_sizes[8];
  const int G = NGRAPH;

  char* base = (char*)d_ws;
  size_t off = 0;
  auto alloc = [&](size_t bytes) -> void* {
    void* p = base + off;
    off += (bytes + 255) & ~(size_t)255;
    return p;
  };
  float*  bufA  = (float*)alloc((size_t)N * DP * 4);
  float*  bufB  = (float*)alloc((size_t)N * DP * 4);
  __bf16* bufAh = (__bf16*)alloc((size_t)N * DP * 2);   // bf16 A-operand scratch
  float*  dinv  = (float*)alloc((size_t)N * 4);
  __bf16* Wh    = (__bf16*)alloc((size_t)7 * DP * DP * 2);
  float*  bp    = (float*)alloc((size_t)7 * DP * 4);
  float*  colS  = (float*)alloc((size_t)2 * DP * 4);
  float*  colQ  = colS + DP;
  float*  gA    = (float*)alloc((size_t)G * DP * 4);
  float*  gB    = (float*)alloc((size_t)G * DP * 4);
  float*  jA    = (float*)alloc((size_t)J * DP * 4);
  float*  jB    = (float*)alloc((size_t)J * DP * 4);
  float*  f0    = (float*)alloc((size_t)G * DP * 4);
  float*  f1    = (float*)alloc((size_t)G * DP * 4);
  __bf16* f0h   = (__bf16*)alloc((size_t)G * DP * 2);
  __bf16* f1h   = (__bf16*)alloc((size_t)G * DP * 2);
  float*  cnt   = (float*)alloc((size_t)J * 4);

  auto blk = [](long n) { return (unsigned)((n + 255) / 256); };
  auto cvt = [&](const float* s, __bf16* d, long elems) {
    f32tobf16_kernel<<<blk(elems / 8), 256, 0, stream>>>(s, d, elems / 8);
  };

  // Pad weights straight to bf16 (zeros in rows/cols 300..319), biases to f32.
  pad2d_bf16_kernel<<<blk(5L * DP * DP), 256, 0, stream>>>(gnn_W, Wh, DC, DC, DP, DP, 5L * DP * DP);
  pad2d_bf16_kernel<<<blk(2L * DP * DP), 256, 0, stream>>>(jct_Wsrc, Wh + 5L * DP * DP, DC, DC, DP, DP, 2L * DP * DP);
  pad2d_kernel<<<blk(5L * DP), 256, 0, stream>>>(gnn_b, bp, 1, DC, 1, DP, 5L * DP);
  pad2d_kernel<<<blk(2L * DP), 256, 0, stream>>>(jct_bsrc, bp + 5L * DP, 1, DC, 1, DP, 2L * DP);

  const long t80N = (long)N * (DP / 4);
  const long t75N = (long)N * (DC / 4);

  auto run_gcn = [&](const int* x, const int* ei, const int* ea) {
    embed_kernel<<<blk(t80N), 256, 0, stream>>>(x, ae1, ae2, bufA, N);
    fill_kernel<<<blk(N), 256, 0, stream>>>(dinv, 1.0f, N);       // self-loop degree
    deg_kernel<<<blk(E), 256, 0, stream>>>(ei, dinv, E);
    rsqrt_kernel<<<blk(N), 256, 0, stream>>>(dinv, N);
    for (int l = 0; l < 5; ++l) {
      const __bf16* W  = Wh + (size_t)l * DP * DP;
      const float*  bb = bp + (size_t)l * DP;
      const float*  e1 = gnn_ee1 + (size_t)l * 6 * DC;
      const float*  e2 = gnn_ee2 + (size_t)l * 3 * DC;
      cvt(bufA, bufAh, (long)N * DP);
      dim3 gg((N + TM - 1) / TM, DP / TN);
      gemm_wmma_kernel<<<gg, 256, 0, stream>>>(bufAh, DP, W, DP, bufB, DP, bb, 1.0f, N, DP);
      self_init_kernel<<<blk(t80N), 256, 0, stream>>>(bufB, e1 + 4 * DC, e2, dinv, bufA, N);
      edge_msg_kernel<<<blk((long)E * 32), 256, 0, stream>>>(ei, ea, bufB, e1, e2, dinv, bufA, E);
      fill_kernel<<<blk(2L * DP), 256, 0, stream>>>(colS, 0.0f, 2L * DP);
      colstat_kernel<<<dim3(DC, 32), 256, 0, stream>>>(bufA, colS, colQ, N);
      bn_apply_kernel<<<blk(t75N), 256, 0, stream>>>(bufA, colS, colQ,
          bn_gamma + (size_t)l * DC, bn_beta + (size_t)l * DC, N, l < 4 ? 1 : 0);
    }
  };

  auto run_jct = [&](float* inb, float* tmpb, int M) {   // result ends in inb
    dim3 gg((M + TM - 1) / TM, DP / TN);
    long t75 = (long)M * (DC / 4);
    cvt(inb, bufAh, (long)M * DP);
    gemm_wmma_kernel<<<gg, 256, 0, stream>>>(bufAh, DP, Wh + 5L * DP * DP, DP, tmpb, DP, bp + 5L * DP, 1.0f, M, DP);
    addse_kernel<<<blk(t75), 256, 0, stream>>>(tmpb, jct_ee1 + 4 * DC, jct_ee2, M, 1);
    cvt(tmpb, bufAh, (long)M * DP);
    gemm_wmma_kernel<<<gg, 256, 0, stream>>>(bufAh, DP, Wh + 6L * DP * DP, DP, inb, DP, bp + 6L * DP, 1.0f, M, DP);
    addse_kernel<<<blk(t75), 256, 0, stream>>>(inb, jct_ee1 + 10 * DC, jct_ee2 + 3 * DC, M, 0);
  };

  // ---------------- branch 0: whole molecules ----------------
  run_gcn(batch_x, batch_ei, batch_ea);
  fill_kernel<<<blk((long)G * DP), 256, 0, stream>>>(gA, 0.0f, (long)G * DP);
  fill_kernel<<<blk(G), 256, 0, stream>>>(cnt, 0.0f, G);
  pool_scatter_kernel<<<blk((long)N * 32), 256, 0, stream>>>(bufA, batch_gid, gA, cnt, N);
  pool_div_kernel<<<blk((long)G * (DP / 4)), 256, 0, stream>>>(gA, cnt, G);
  run_jct(gA, gB, G);
  l2norm_kernel<<<blk((long)G * 32), 256, 0, stream>>>(gA, f0, G);

  // ---------------- branch 1: fragments -> junctions ----------------
  run_gcn(frag_x, frag_ei, frag_ea);
  fill_kernel<<<blk((long)J * DP), 256, 0, stream>>>(jA, 0.0f, (long)J * DP);
  fill_kernel<<<blk(J), 256, 0, stream>>>(cnt, 0.0f, J);
  pool_scatter_kernel<<<blk((long)N * 32), 256, 0, stream>>>(bufA, frag_jid, jA, cnt, N);
  pool_div_kernel<<<blk((long)J * (DP / 4)), 256, 0, stream>>>(jA, cnt, J);
  mask_kernel<<<blk((long)J * (DC / 4)), 256, 0, stream>>>(jA, jct_mask, mask_emb, J);
  run_jct(jA, jB, J);
  fill_kernel<<<blk((long)G * DP), 256, 0, stream>>>(gB, 0.0f, (long)G * DP);
  fill_kernel<<<blk(G), 256, 0, stream>>>(cnt, 0.0f, G);
  pool_scatter_kernel<<<blk((long)J * 32), 256, 0, stream>>>(jA, jct_gid, gB, cnt, J);
  pool_div_kernel<<<blk((long)G * (DP / 4)), 256, 0, stream>>>(gB, cnt, G);
  l2norm_kernel<<<blk((long)G * 32), 256, 0, stream>>>(gB, f1, G);

  // ---------------- logits = (f0 @ f1^T) / TEMP ----------------
  cvt(f0, f0h, (long)G * DP);
  cvt(f1, f1h, (long)G * DP);
  dim3 gl((G + TM - 1) / TM, (G + TN - 1) / TN);
  gemm_wmma_kernel<<<gl, 256, 0, stream>>>(f0h, DP, f1h, DP, (float*)d_out, G,
                                           nullptr, 25.0f, G, G);
}
